// PointTransformerLayer_42563125903627
// MI455X (gfx1250) — matrix operand order repaired and checked by reading, not compiled, not run on hardware
//
#include <hip/hip_runtime.h>
#include <hip/hip_bf16.h>
#include <hip/hip_fp16.h>

typedef __attribute__((ext_vector_type(16))) _Float16 v16h;
typedef __attribute__((ext_vector_type(8)))  _Float16 v8h;
typedef __attribute__((ext_vector_type(2)))  _Float16 v2h;
typedef __attribute__((ext_vector_type(8)))  float    v8f;
typedef __attribute__((ext_vector_type(4)))  unsigned int u32x4;
typedef __attribute__((ext_vector_type(8)))  unsigned int u32x8;

#define NSAMPLE 16
#define CH      256
#define INNER   32
#define SHAREF  8

__device__ __forceinline__ float wave_reduce_add(float v) {
#pragma unroll
  for (int o = 16; o > 0; o >>= 1) v += __shfl_xor(v, o, 32);
  return v;
}

// ---------------------------------------------------------------------------
// TDM: DMA a 2D tile (tile_x elems x tile_y rows) from global to LDS.
// ds_code: 0=1B, 1=2B, 2=4B element size. Tracked by TENSORcnt.
// ---------------------------------------------------------------------------
__device__ __forceinline__ void tdm_load_tile(unsigned lds_off, const void* gptr,
                                              unsigned ds_code, unsigned tile_x,
                                              unsigned tile_y, unsigned row_stride_elems) {
  unsigned long long ga = (unsigned long long)gptr;
  u32x4 g0;
  g0[0] = 1u;                                                 // count=1, user D#
  g0[1] = lds_off;                                            // lds_addr (bytes)
  g0[2] = (unsigned)ga;                                       // global_addr[31:0]
  g0[3] = (unsigned)((ga >> 32) & 0x01FFFFFFu) | 0x80000000u; // addr[56:32]|type=2
  const unsigned TD0 = 0x40000000u, TD1 = 0x40000000u;        // huge: never OOB
  u32x8 g1;
  g1[0] = ds_code << 16;                                      // data_size
  g1[1] = (TD0 & 0xFFFFu) << 16;
  g1[2] = ((TD0 >> 16) & 0xFFFFu) | ((TD1 & 0xFFFFu) << 16);
  g1[3] = ((TD1 >> 16) & 0xFFFFu) | (tile_x << 16);           // tile_dim0
  g1[4] = tile_y;                                             // tile_dim1
  g1[5] = row_stride_elems;                                   // dim0_stride lo
  g1[6] = 0u;
  g1[7] = 0u;
  asm volatile("tensor_load_to_lds %0, %1" :: "s"(g0), "s"(g1) : "memory");
}

__device__ __forceinline__ unsigned lds_off_of(const void* p) {
  return (unsigned)(unsigned long long)p;                     // low 32b = LDS offset
}

// ---------------------------------------------------------------------------
// 0a. zero stats accumulators
// ---------------------------------------------------------------------------
__global__ void zero_kernel(float* __restrict__ p, int cnt) {
  int i = blockIdx.x * blockDim.x + threadIdx.x;
  if (i < cnt) p[i] = 0.0f;
}

// 0b. convert f32 -> f16 (activations)
__global__ void cvt_f16_kernel(const float* __restrict__ X, _Float16* __restrict__ XH, int total) {
  int i = blockIdx.x * 256 + threadIdx.x;
  if (i < total) XH[i] = (_Float16)X[i];
}

// 0c. pack weight W[K,N] (f32 row-major) into WMMA B-fragment order (f16)
__global__ void pack_w_kernel(const float* __restrict__ W, _Float16* __restrict__ WF,
                              int K, int N) {
  const int gid = blockIdx.x * 256 + threadIdx.x;
  const int total = (K / 32) * (N / 16) * 32;
  if (gid >= total) return;
  const int l = gid & 31;
  const int tile = gid >> 5;
  const int nt = N >> 4;
  const int kt = tile / nt, ct = tile % nt;
  const int half = l >> 4, lm = l & 15;
  _Float16* dst = WF + (size_t)tile * 512 + l * 16;
#pragma unroll
  for (int e = 0; e < 16; ++e) {
    const int k = kt * 32 + half * 8 + (e & 7) + (e >> 3) * 16;
    dst[e] = (_Float16)W[(size_t)k * N + ct * 16 + lm];
  }
}

// ---------------------------------------------------------------------------
// 1. brute-force kNN (k=16, self included) per point cloud, LDS-tiled
// ---------------------------------------------------------------------------
__global__ void knn_kernel(const float* __restrict__ P, int* __restrict__ IDX, int NB) {
  __shared__ float sx[256], sy[256], sz[256];
  const int bpb = NB / 256;
  const int b   = blockIdx.x / bpb;
  const int q   = (blockIdx.x % bpb) * 256 + threadIdx.x;
  const int gq  = b * NB + q;
  const float qx = P[gq*3+0], qy = P[gq*3+1], qz = P[gq*3+2];
  float bd[NSAMPLE]; int bi[NSAMPLE];
#pragma unroll
  for (int t = 0; t < NSAMPLE; ++t) { bd[t] = 3.0e38f; bi[t] = gq; }
  for (int c0 = 0; c0 < NB; c0 += 256) {
    const int gc = b * NB + c0 + threadIdx.x;
    sx[threadIdx.x] = P[gc*3+0];
    sy[threadIdx.x] = P[gc*3+1];
    sz[threadIdx.x] = P[gc*3+2];
    __syncthreads();
    for (int j = 0; j < 256; ++j) {
      const float dx = qx - sx[j], dy = qy - sy[j], dz = qz - sz[j];
      const float d  = dx*dx + dy*dy + dz*dz;
      if (d < bd[NSAMPLE-1]) {
        int k = NSAMPLE - 1;
        while (k > 0 && bd[k-1] > d) { bd[k] = bd[k-1]; bi[k] = bi[k-1]; --k; }
        bd[k] = d; bi[k] = b * NB + c0 + j;
      }
    }
    __syncthreads();
  }
#pragma unroll
  for (int t = 0; t < NSAMPLE; ++t) IDX[gq*NSAMPLE + t] = bi[t];
}

// ---------------------------------------------------------------------------
// 2. fused QKV: Yq/Yk/Yv[M,256] = XH[M,256] @ Wq/k/v + bias.
//    Block = 128M x 64N; one TDM-staged f16 X slab feeds 12 WMMAs per k-step.
//    B fragments loaded in groups of 4 (batched clause) per weight matrix.
// ---------------------------------------------------------------------------
__global__ void __launch_bounds__(256, 1)
qkv_gemm_kernel(const _Float16* __restrict__ XH,
                const _Float16* __restrict__ WFQ,
                const _Float16* __restrict__ WFK,
                const _Float16* __restrict__ WFV,
                const float* __restrict__ BQ, const float* __restrict__ BK,
                const float* __restrict__ BV,
                float* __restrict__ YQ, float* __restrict__ YK,
                float* __restrict__ YV) {
  __shared__ _Float16 sXh[2][128 * 32];                      // 2 x 8 KB
  const int lane = threadIdx.x & 31, wid = threadIdx.x >> 5;
  const int bm = blockIdx.x >> 2, bn = blockIdx.x & 3;       // 4 N-blocks of 64
  const int row0 = bm * 128 + wid * 16;
  const int colb = bn * 64, ctb = bn * 4;
  const int half = lane >> 4, lm = lane & 15;
  const _Float16* Xblk = XH + (size_t)(bm * 128) * CH;
  const _Float16* const WF[3] = {WFQ, WFK, WFV};
  const float* const BIAS[3]  = {BQ, BK, BV};
  float* const Y[3]           = {YQ, YK, YV};

  v8f acc[3][4] = {};

  if (wid == 0)
    tdm_load_tile(lds_off_of(&sXh[0][0]), Xblk, 1u, 32u, 128u, (unsigned)CH);

#pragma unroll 1
  for (int i = 0; i < 8; ++i) {                              // k0 = i*32
    if (wid == 0) {
      if (i + 1 < 8) {
        tdm_load_tile(lds_off_of(&sXh[(i + 1) & 1][0]), Xblk + (i + 1) * 32,
                      1u, 32u, 128u, (unsigned)CH);
        __builtin_amdgcn_s_wait_tensorcnt(1);                // slab i complete
      } else {
        __builtin_amdgcn_s_wait_tensorcnt(0);
      }
    }
    __syncthreads();
    const _Float16* xs = &sXh[i & 1][(wid * 16 + lm) * 32];
    const v8h lo = *(const v8h*)(xs + half * 8);
    const v8h hi = *(const v8h*)(xs + 16 + half * 8);
    const v16h a = __builtin_shufflevector(lo, hi, 0, 1, 2, 3, 4, 5, 6, 7,
                                           8, 9, 10, 11, 12, 13, 14, 15);
#pragma unroll
    for (int w = 0; w < 3; ++w) {
      v16h bf[4];                                            // batched fragment group
#pragma unroll
      for (int nt = 0; nt < 4; ++nt)
        bf[nt] = *(const v16h*)(WF[w] + (size_t)(i * 16 + ctb + nt) * 512 + lane * 16);
#pragma unroll
      for (int nt = 0; nt < 4; ++nt)
        acc[w][nt] = __builtin_amdgcn_wmma_f32_16x16x32_f16(false, a, false, bf[nt],
                                                            (short)0, acc[w][nt], false, false);
    }
    __syncthreads();
  }
#pragma unroll
  for (int w = 0; w < 3; ++w) {
#pragma unroll
    for (int nt = 0; nt < 4; ++nt) {
      const float bs = BIAS[w][colb + nt*16 + lm];
#pragma unroll
      for (int r = 0; r < 8; ++r)                            // C: (M=r+8*half, N=lm)
        Y[w][(size_t)(row0 + r + half*8) * CH + colb + nt*16 + lm] = acc[w][nt][r] + bs;
    }
  }
}

// ---------------------------------------------------------------------------
// 3. h = (p[idx]-p) @ wp1 + bp1  [rows,3]  + bn_p channel stats
// ---------------------------------------------------------------------------
__global__ void hstat_kernel(const float* __restrict__ P, const int* __restrict__ IDX,
                             const float* __restrict__ wp1, const float* __restrict__ bp1,
                             float* __restrict__ H,
                             float* __restrict__ sp_sum, float* __restrict__ sp_sq) {
  const int gid = blockIdx.x * 256 + threadIdx.x;
  const int pt  = gid >> 4;
  const int nb  = IDX[gid];
  const float rx = P[nb*3+0] - P[pt*3+0];
  const float ry = P[nb*3+1] - P[pt*3+1];
  const float rz = P[nb*3+2] - P[pt*3+2];
  float h[3];
#pragma unroll
  for (int j = 0; j < 3; ++j) {
    h[j] = rx*wp1[0*3+j] + ry*wp1[1*3+j] + rz*wp1[2*3+j] + bp1[j];
    H[(size_t)gid*3 + j] = h[j];
  }
#pragma unroll
  for (int j = 0; j < 3; ++j) {
    const float s = wave_reduce_add(h[j]);
    const float q = wave_reduce_add(h[j]*h[j]);
    if ((threadIdx.x & 31) == 0) { atomicAdd(&sp_sum[j], s); atomicAdd(&sp_sq[j], q); }
  }
}

// ---------------------------------------------------------------------------
// 4. BN finalize: scale = g*rstd, shift = b - mean*g*rstd
// ---------------------------------------------------------------------------
__global__ void bn_finalize_kernel(const float* __restrict__ sum, const float* __restrict__ sumsq,
                                   const float* __restrict__ g, const float* __restrict__ bta,
                                   float* __restrict__ scale, float* __restrict__ shift,
                                   int nch, float invcnt) {
  const int c = blockIdx.x * blockDim.x + threadIdx.x;
  if (c < nch) {
    const float m  = sum[c] * invcnt;
    const float v  = sumsq[c] * invcnt - m * m;
    const float rs = rsqrtf(v + 1e-5f);
    scale[c] = g[c] * rs;
    shift[c] = bta[c] - m * g[c] * rs;
  }
}

// ---------------------------------------------------------------------------
// 5. r_qk = x_k[idx] - x_q + p_e   (f16, 2 channels/thread)  + bn_w1 stats
// ---------------------------------------------------------------------------
__global__ void rqk_kernel(const float* __restrict__ XK, const float* __restrict__ XQ,
                           const int* __restrict__ IDX, const float* __restrict__ H,
                           const float* __restrict__ psc, const float* __restrict__ psh,
                           const float* __restrict__ wp2, const float* __restrict__ bp2,
                           _Float16* __restrict__ RQK,
                           float* __restrict__ s1_sum, float* __restrict__ s1_sq) {
  const int pt = blockIdx.x;
  const int c  = threadIdx.x * 2;                  // 128 threads x 2 channels
  __shared__ float hb[NSAMPLE * 3];
  if (threadIdx.x < NSAMPLE) {
    const int t = threadIdx.x;
    const float* hh = H + (size_t)(pt * NSAMPLE + t) * 3;
#pragma unroll
    for (int j = 0; j < 3; ++j)
      hb[t*3 + j] = fmaxf(hh[j] * psc[j] + psh[j], 0.0f);
  }
  __syncthreads();
  const float2 xq = *(const float2*)(XQ + (size_t)pt * CH + c);
  const float w0a = wp2[0*CH + c], w1a = wp2[1*CH + c], w2a = wp2[2*CH + c];
  const float w0b = wp2[0*CH + c+1], w1b = wp2[1*CH + c+1], w2b = wp2[2*CH + c+1];
  const float bba = bp2[c], bbb = bp2[c+1];
  float suma = 0.0f, sqa = 0.0f, sumb = 0.0f, sqb = 0.0f;
  for (int t = 0; t < NSAMPLE; ++t) {
    const int nb = IDX[pt*NSAMPLE + t];
    const float h0 = hb[t*3+0], h1 = hb[t*3+1], h2 = hb[t*3+2];
    const float2 xk = *(const float2*)(XK + (size_t)nb * CH + c);
    const float ra = xk.x - xq.x + (h0*w0a + h1*w1a + h2*w2a + bba);
    const float rb = xk.y - xq.y + (h0*w0b + h1*w1b + h2*w2b + bbb);
    v2h pk; pk[0] = (_Float16)ra; pk[1] = (_Float16)rb;
    *(v2h*)(RQK + (size_t)(pt*NSAMPLE + t) * CH + c) = pk;
    suma += ra; sqa += ra * ra;
    sumb += rb; sqb += rb * rb;
  }
  atomicAdd(&s1_sum[c], suma);   atomicAdd(&s1_sq[c], sqa);
  atomicAdd(&s1_sum[c+1], sumb); atomicAdd(&s1_sq[c+1], sqb);
}

// ---------------------------------------------------------------------------
// 6. W1[rows,32] = relu(bn(r_qk)) @ ww1 + bw1   (2 WMMAs/wave) + bn_w2 stats
// ---------------------------------------------------------------------------
__global__ void w1_gemm_kernel(const _Float16* __restrict__ RQK,
                               const float* __restrict__ sc, const float* __restrict__ sh,
                               const _Float16* __restrict__ WF1, const float* __restrict__ bw1,
                               float* __restrict__ W1,
                               float* __restrict__ s2_sum, float* __restrict__ s2_sq) {
  const int lane = threadIdx.x & 31, wid = threadIdx.x >> 5;
  const int tile = blockIdx.x * 8 + wid;
  const int row0 = tile * 16;
  const int half = lane >> 4, lm = lane & 15;
  v8f acc0 = {}, acc1 = {};
  const _Float16* rr = RQK + (size_t)(row0 + lm) * CH;
  for (int kt = 0; kt < 8; ++kt) {
    const int k0 = kt * 32;
    const v16h b0 = *(const v16h*)(WF1 + (size_t)(kt * 2 + 0) * 512 + lane * 16);
    const v16h b1 = *(const v16h*)(WF1 + (size_t)(kt * 2 + 1) * 512 + lane * 16);
    v16h a;
#pragma unroll
    for (int e = 0; e < 8; ++e) {
      const int k1 = k0 + half*8 + e;
      const int k2 = k0 + 16 + half*8 + e;
      a[e]     = (_Float16)fmaxf((float)rr[k1] * sc[k1] + sh[k1], 0.0f);
      a[e + 8] = (_Float16)fmaxf((float)rr[k2] * sc[k2] + sh[k2], 0.0f);
    }
    acc0 = __builtin_amdgcn_wmma_f32_16x16x32_f16(false, a, false, b0, (short)0, acc0, false, false);
    acc1 = __builtin_amdgcn_wmma_f32_16x16x32_f16(false, a, false, b1, (short)0, acc1, false, false);
  }
  const float bs0 = bw1[lm], bs1 = bw1[16 + lm];
  float ls0 = 0.0f, lq0 = 0.0f, ls1 = 0.0f, lq1 = 0.0f;
#pragma unroll
  for (int r = 0; r < 8; ++r) {
    const float v0 = acc0[r] + bs0;
    const float v1 = acc1[r] + bs1;
    W1[(size_t)(row0 + r + half*8) * INNER + lm]      = v0;
    W1[(size_t)(row0 + r + half*8) * INNER + 16 + lm] = v1;
    ls0 += v0; lq0 += v0 * v0;
    ls1 += v1; lq1 += v1 * v1;
  }
  atomicAdd(&s2_sum[lm], ls0);      atomicAdd(&s2_sq[lm], lq0);
  atomicAdd(&s2_sum[16 + lm], ls1); atomicAdd(&s2_sq[16 + lm], lq1);
}

// ---------------------------------------------------------------------------
// 7. final: scores = relu(bn(W1)) @ ww2 + bw2 (2 WMMAs per point),
//    softmax over 16 neighbors, out = sum_t (x_v[idx]+p_e) * w
// ---------------------------------------------------------------------------
__global__ void out_kernel(const float* __restrict__ W1,
                           const float* __restrict__ sc2, const float* __restrict__ sh2,
                           const _Float16* __restrict__ WF2, const float* __restrict__ bw2,
                           const float* __restrict__ XV, const int* __restrict__ IDX,
                           const float* __restrict__ H,
                           const float* __restrict__ psc, const float* __restrict__ psh,
                           const float* __restrict__ wp2, const float* __restrict__ bp2,
                           float* __restrict__ OUT) {
  __shared__ float lw[8][NSAMPLE * INNER];
  __shared__ float lhb[8][NSAMPLE * 3];
  const int lane = threadIdx.x & 31, wid = threadIdx.x >> 5;
  const int pt = blockIdx.x * 8 + wid;
  const int half = lane >> 4, lm = lane & 15;

  const float* w1p = W1 + (size_t)pt * NSAMPLE * INNER;
  v16h a;
#pragma unroll
  for (int e = 0; e < 8; ++e) {
    const int k1 = half*8 + e;
    const int k2 = 16 + half*8 + e;
    a[e]     = (_Float16)fmaxf(w1p[lm*INNER + k1] * sc2[k1] + sh2[k1], 0.0f);
    a[e + 8] = (_Float16)fmaxf(w1p[lm*INNER + k2] * sc2[k2] + sh2[k2], 0.0f);
  }
  const v16h b0 = *(const v16h*)(WF2 + (size_t)0 * 512 + lane * 16);
  const v16h b1 = *(const v16h*)(WF2 + (size_t)1 * 512 + lane * 16);
  v8f c0 = {}, c1 = {};
  c0 = __builtin_amdgcn_wmma_f32_16x16x32_f16(false, a, false, b0, (short)0, c0, false, false);
  c1 = __builtin_amdgcn_wmma_f32_16x16x32_f16(false, a, false, b1, (short)0, c1, false, false);

  const float bb0 = bw2[lm], bb1 = bw2[16 + lm];
  float m0 = -3.0e38f, m1 = -3.0e38f;
#pragma unroll
  for (int r = 0; r < 8; ++r) {
    c0[r] += bb0; c1[r] += bb1;
    m0 = fmaxf(m0, c0[r]); m1 = fmaxf(m1, c1[r]);
  }
  m0 = fmaxf(m0, __shfl_xor(m0, 16, 32));
  m1 = fmaxf(m1, __shfl_xor(m1, 16, 32));
  float s0 = 0.0f, s1 = 0.0f;
#pragma unroll
  for (int r = 0; r < 8; ++r) {
    c0[r] = __expf(c0[r] - m0); s0 += c0[r];
    c1[r] = __expf(c1[r] - m1); s1 += c1[r];
  }
  s0 += __shfl_xor(s0, 16, 32);
  s1 += __shfl_xor(s1, 16, 32);
  const float i0 = 1.0f / s0, i1 = 1.0f / s1;
#pragma unroll
  for (int r = 0; r < 8; ++r) {
    lw[wid][(r + half*8)*INNER + lm]      = c0[r] * i0;
    lw[wid][(r + half*8)*INNER + 16 + lm] = c1[r] * i1;
  }
  if (lane < NSAMPLE) {
    const float* hh = H + (size_t)(pt * NSAMPLE + lane) * 3;
#pragma unroll
    for (int j = 0; j < 3; ++j)
      lhb[wid][lane*3 + j] = fmaxf(hh[j] * psc[j] + psh[j], 0.0f);
  }
  __syncthreads();

  const int i = lane;
  float accs[SHAREF];
#pragma unroll
  for (int s = 0; s < SHAREF; ++s) accs[s] = 0.0f;
  for (int t = 0; t < NSAMPLE; ++t) {
    const int nb = IDX[pt*NSAMPLE + t];
    const float h0 = lhb[wid][t*3+0], h1 = lhb[wid][t*3+1], h2 = lhb[wid][t*3+2];
    const float wgt = lw[wid][t*INNER + i];
    const float* xvr = XV + (size_t)nb * CH;
#pragma unroll
    for (int s = 0; s < SHAREF; ++s) {
      const int ch = s*INNER + i;
      const float pe = h0*wp2[0*CH + ch] + h1*wp2[1*CH + ch] + h2*wp2[2*CH + ch] + bp2[ch];
      accs[s] += (xvr[ch] + pe) * wgt;
    }
  }
#pragma unroll
  for (int s = 0; s < SHAREF; ++s)
    OUT[(size_t)pt * CH + s*INNER + i] = accs[s];
}

// ---------------------------------------------------------------------------
extern "C" void kernel_launch(void* const* d_in, const int* in_sizes, int n_in,
                              void* d_out, int out_size, void* d_ws, size_t ws_size,
                              hipStream_t stream) {
  (void)n_in; (void)out_size; (void)ws_size;
  const float* P    = (const float*)d_in[0];
  const float* X    = (const float*)d_in[1];
  const float* WQ   = (const float*)d_in[3];  const float* BQ  = (const float*)d_in[4];
  const float* WK   = (const float*)d_in[5];  const float* BK  = (const float*)d_in[6];
  const float* WV   = (const float*)d_in[7];  const float* BV  = (const float*)d_in[8];
  const float* WP1  = (const float*)d_in[9];  const float* BP1 = (const float*)d_in[10];
  const float* BNPG = (const float*)d_in[11]; const float* BNPB= (const float*)d_in[12];
  const float* WP2  = (const float*)d_in[13]; const float* BP2 = (const float*)d_in[14];
  const float* BN1G = (const float*)d_in[15]; const float* BN1B= (const float*)d_in[16];
  const float* WW1  = (const float*)d_in[17]; const float* BW1 = (const float*)d_in[18];
  const float* BN2G = (const float*)d_in[19]; const float* BN2B= (const float*)d_in[20];
  const float* WW2  = (const float*)d_in[21]; const float* BW2 = (const float*)d_in[22];

  const int n    = in_sizes[1] / CH;       // 16384
  const int B    = in_sizes[2];            // 4
  const int NB   = n / B;                  // 4096
  const int rows = n * NSAMPLE;            // 262144
  const float invcnt = 1.0f / (float)rows;

  char* ws = (char*)d_ws;
  float* stats  = (float*)ws;
  float* sp_sum = stats + 0;    float* sp_sq = stats + 4;
  float* s1_sum = stats + 8;    float* s1_sq = stats + 264;
  float* s2_sum = stats + 520;  float* s2_sq = stats + 552;
  float* p_sc   = stats + 1024; float* p_sh  = stats + 1028;
  float* w1_sc  = stats + 1032; float* w1_sh = stats + 1288;
  float* w2_sc  = stats + 1544; float* w2_sh = stats + 1576;
  size_t off = 8192;
  int*      IDX = (int*)(ws + off);       off += (size_t)rows * sizeof(int);
  float*    XQ  = (float*)(ws + off);     off += (size_t)n * CH * sizeof(float);
  float*    XK  = (float*)(ws + off);     off += (size_t)n * CH * sizeof(float);
  float*    XV  = (float*)(ws + off);     off += (size_t)n * CH * sizeof(float);
  float*    H   = (float*)(ws + off);     off += (size_t)rows * 3 * sizeof(float);
  _Float16* RQK = (_Float16*)(ws + off);  off += (size_t)rows * CH * sizeof(_Float16);
  float*    W1  = (float*)(ws + off);     off += (size_t)rows * INNER * sizeof(float);
  _Float16* XH  = (_Float16*)(ws + off);  off += (size_t)n * CH * sizeof(_Float16);
  _Float16* WFQ = (_Float16*)(ws + off);  off += (size_t)(CH/32)*(CH/16)*512 * sizeof(_Float16);
  _Float16* WFK = (_Float16*)(ws + off);  off += (size_t)(CH/32)*(CH/16)*512 * sizeof(_Float16);
  _Float16* WFV = (_Float16*)(ws + off);  off += (size_t)(CH/32)*(CH/16)*512 * sizeof(_Float16);
  _Float16* WF1 = (_Float16*)(ws + off);  off += (size_t)(CH/32)*(INNER/16)*512 * sizeof(_Float16);
  _Float16* WF2 = (_Float16*)(ws + off);  off += (size_t)(INNER/32)*(INNER/16)*512 * sizeof(_Float16);

  zero_kernel<<<4, 256, 0, stream>>>(stats, 1024);

  // prep: f16 activations + fragment-packed f16 weights
  cvt_f16_kernel<<<(n * CH) / 256, 256, 0, stream>>>(X, XH, n * CH);
  pack_w_kernel<<<16, 256, 0, stream>>>(WQ,  WFQ, CH, CH);
  pack_w_kernel<<<16, 256, 0, stream>>>(WK,  WFK, CH, CH);
  pack_w_kernel<<<16, 256, 0, stream>>>(WV,  WFV, CH, CH);
  pack_w_kernel<<<2, 256, 0, stream>>>(WW1, WF1, CH, INNER);
  pack_w_kernel<<<1, 256, 0, stream>>>(WW2, WF2, INNER, INNER);

  knn_kernel<<<B * (NB / 256), 256, 0, stream>>>(P, IDX, NB);

  // fused QKV projections (WMMA + TDM-staged A slabs)
  qkv_gemm_kernel<<<(n / 128) * (CH / 64), 256, 0, stream>>>(XH, WFQ, WFK, WFV,
                                                             BQ, BK, BV, XQ, XK, XV);

  hstat_kernel<<<rows / 256, 256, 0, stream>>>(P, IDX, WP1, BP1, H, sp_sum, sp_sq);
  bn_finalize_kernel<<<1, 64, 0, stream>>>(sp_sum, sp_sq, BNPG, BNPB, p_sc, p_sh, 3, invcnt);

  rqk_kernel<<<n, 128, 0, stream>>>(XK, XQ, IDX, H, p_sc, p_sh, WP2, BP2, RQK, s1_sum, s1_sq);
  bn_finalize_kernel<<<1, 256, 0, stream>>>(s1_sum, s1_sq, BN1G, BN1B, w1_sc, w1_sh, CH, invcnt);

  w1_gemm_kernel<<<(rows / 16) / 8, 256, 0, stream>>>(RQK, w1_sc, w1_sh, WF1, BW1,
                                                      W1, s2_sum, s2_sq);
  bn_finalize_kernel<<<1, 64, 0, stream>>>(s2_sum, s2_sq, BN2G, BN2B, w2_sc, w2_sh, INNER, invcnt);

  out_kernel<<<n / 8, 256, 0, stream>>>(W1, w2_sc, w2_sh, WF2, BW2, XV, IDX, H,
                                        p_sc, p_sh, WP2, BP2, (float*)d_out);
}
